// simple_pydict_Encoding_50646254355199
// MI455X (gfx1250) — matrix operand order repaired and checked by reading, not compiled, not run on hardware
//
#include <hip/hip_runtime.h>
#include <hip/hip_bf16.h>
#include <stdint.h>

// ---------------------------------------------------------------------------
// CDNA5 (gfx1250, wave32) fully-fused Encoding kernel.
//   out = X + E1(X) + E2(X2) + E3(X3),  X2 = relu(bn1(W1@X)), X3 = relu(bn2(W2@X2))
// All matmuls via v_wmma_f32_16x16x32_bf16. One block = 64 spatial columns,
// full D=512, everything staged through LDS (X2/X3 never hit HBM).
// Weight tiles streamed with global_load_async_to_lds_b128 (ASYNCcnt),
// double-buffered so the copy of K-slab k+1 overlaps WMMA work on slab k.
// ---------------------------------------------------------------------------

typedef __attribute__((ext_vector_type(16))) __bf16 v16bf;
typedef __attribute__((ext_vector_type(8)))  float  v8f;
typedef unsigned short us;

#define DCH   512
#define NSP   9216          // 96*96
#define NT    64            // columns per block
#define SROW  520           // LDS activation row stride (u16), 1040B, mult of 16
#define WROW  40            // LDS weight row stride (u16), 80B, mult of 16
#define LROW  68            // LDS logits row stride (f32), 272B, mult of 16
#define AROW  72            // LDS softmax row stride (u16), 144B, mult of 16
#define EPSV  1e-5f

union AFrag { v16bf bf; unsigned u[8]; uint4 q[2]; };

__device__ __forceinline__ us f2bf(float f) {
  unsigned u = __float_as_uint(f);
  u += 0x7FFFu + ((u >> 16) & 1u);          // round-to-nearest-even
  return (us)(u >> 16);
}

__device__ __forceinline__ v8f v8zero() {
  v8f z;
#pragma unroll
  for (int i = 0; i < 8; ++i) z[i] = 0.f;
  return z;
}

__device__ __forceinline__ v8f wmma_bf16(const AFrag& a, const AFrag& b, v8f c) {
  // (neg_a, A, neg_b, B, c_mod, C, reuse_a, reuse_b)
  return __builtin_amdgcn_wmma_f32_16x16x32_bf16(false, a.bf, false, b.bf,
                                                 (short)0, c, false, false);
}

// B-fragment (32x16, K x N) from transposed-[n][d] LDS tile: two b128 loads.
__device__ __forceinline__ void load_bfrag_S(AFrag& f, const us* S, int n,
                                             int k0, int lh) {
  const uint4* p = (const uint4*)(S + n * SROW + k0 + lh * 8);
  f.q[0] = p[0];
  f.q[1] = p[2];
}

// A-fragment (16x32, M x K) straight from a row-major bf16 global matrix.
__device__ __forceinline__ void load_afrag_G(AFrag& f, const us* base, int rowlen,
                                             int row, int k0, int lh) {
#pragma unroll
  for (int v = 0; v < 8; ++v) {
    int k = k0 + ((v >> 2) << 4) + (lh << 3) + ((v & 3) << 1);
    f.u[v] = *(const unsigned*)(base + row * rowlen + k);
  }
}

// Async copy of one 512x32 bf16 K-slab of W into an LDS stage buffer.
// 2048 b128 transfers per block -> 8 per thread, 16B aligned both sides.
// Tracked by ASYNCcnt; LDS is written directly (no VGPR round-trip).
__device__ __forceinline__ void async_copy_wtile(const us* __restrict__ Wb, int ks,
                                                 us* Wst, int tid) {
  unsigned lbase = (unsigned)(uintptr_t)(const void*)Wst;   // LDS addr = low 32b
#pragma unroll
  for (int j = 0; j < 8; ++j) {
    int idx = tid + j * 256;
    int e = idx >> 2, q = idx & 3;                          // row, 16B chunk
    unsigned long long gp =
        (unsigned long long)(uintptr_t)(Wb + e * DCH + ks * 32 + q * 8);
    unsigned laddr = lbase + (unsigned)(e * WROW + q * 8) * 2u;
    asm volatile("global_load_async_to_lds_b128 %0, %1, off"
                 :: "v"(laddr), "v"(gp) : "memory");
  }
}

__device__ __forceinline__ void wait_async0() {
  asm volatile("s_wait_asynccnt 0x0" ::: "memory");
}

// ---------------------------------------------------------------------------
// Pre-pass: bf16 weight conversion, cw transpose + zero-pad, BN fold.
// ---------------------------------------------------------------------------
__global__ void prep_kernel(const float* __restrict__ W1, const float* __restrict__ W2,
                            const float* __restrict__ s1, const float* __restrict__ b1,
                            const float* __restrict__ m1, const float* __restrict__ v1,
                            const float* __restrict__ s2, const float* __restrict__ b2,
                            const float* __restrict__ m2, const float* __restrict__ v2,
                            const float* __restrict__ cw1, const float* __restrict__ cw2,
                            const float* __restrict__ cw3,
                            us* W1b, us* W2b,
                            us* cwT1, us* cwT2, us* cwT3,
                            us* cwB1, us* cwB2, us* cwB3,
                            float* sc1, float* sh1, float* sc2, float* sh2) {
  int t = blockIdx.x * blockDim.x + threadIdx.x;
  int st = gridDim.x * blockDim.x;
  for (int i = t; i < DCH * DCH; i += st) { W1b[i] = f2bf(W1[i]); W2b[i] = f2bf(W2[i]); }
  for (int i = t; i < DCH; i += st) {
    float iv1 = rsqrtf(v1[i] + EPSV), iv2 = rsqrtf(v2[i] + EPSV);
    sc1[i] = s1[i] * iv1; sh1[i] = b1[i] - m1[i] * s1[i] * iv1;
    sc2[i] = s2[i] * iv2; sh2[i] = b2[i] - m2[i] * s2[i] * iv2;
  }
  // cwT_j : [Kj][512] (A operand of logits GEMM)
  for (int i = t; i < 16 * DCH; i += st) { int k = i >> 9, d = i & 511; cwT1[i] = f2bf(cw1[d * 16 + k]); }
  for (int i = t; i < 32 * DCH; i += st) { int k = i >> 9, d = i & 511; cwT2[i] = f2bf(cw2[d * 32 + k]); }
  for (int i = t; i < 64 * DCH; i += st) { int k = i >> 9, d = i & 511; cwT3[i] = f2bf(cw3[d * 64 + k]); }
  // cwB_j : [512][Kpad] zero-padded (A operand of E GEMM), Kpad = 32/32/64
  for (int i = t; i < DCH * 32; i += st) { int d = i >> 5, k = i & 31; cwB1[i] = (k < 16) ? f2bf(cw1[d * 16 + k]) : (us)0; }
  for (int i = t; i < DCH * 32; i += st) { int d = i >> 5, k = i & 31; cwB2[i] = f2bf(cw2[d * 32 + k]); }
  for (int i = t; i < DCH * 64; i += st) { int d = i >> 6, k = i & 63; cwB3[i] = f2bf(cw3[d * 64 + k]); }
}

// ---------------------------------------------------------------------------
// scratch = Wb(512x512) @ Scur ; bn+relu ; store bf16 transposed into Snext.
// W K-slabs double-buffered via async LDS copies.
// ---------------------------------------------------------------------------
__device__ __forceinline__ void gemm_bn_relu(const us* __restrict__ Wb,
                                             const us* Scur, us* Snext,
                                             us* Wst0, us* Wst1,
                                             const float* __restrict__ sc,
                                             const float* __restrict__ sh,
                                             int tid, int lane, int wid) {
  const int l15 = lane & 15, lh = lane >> 4;
  const int wave_m = wid & 3, wave_n = wid >> 2;
  us* Wbuf[2] = { Wst0, Wst1 };
  v8f c[8][2];
#pragma unroll
  for (int mt = 0; mt < 8; ++mt) { c[mt][0] = v8zero(); c[mt][1] = v8zero(); }

  async_copy_wtile(Wb, 0, Wbuf[0], tid);     // prologue: slab 0 in flight

  for (int ks = 0; ks < 16; ++ks) {
    wait_async0();                           // my slab-ks transfers landed
    __syncthreads();                         // everyone's landed; ks-1 compute done
    if (ks + 1 < 16)                         // overlap: slab ks+1 -> other buffer
      async_copy_wtile(Wb, ks + 1, Wbuf[(ks + 1) & 1], tid);

    const us* Wcur = Wbuf[ks & 1];
    AFrag bfr[2];
#pragma unroll
    for (int nt = 0; nt < 2; ++nt)
      load_bfrag_S(bfr[nt], Scur, wave_n * 32 + nt * 16 + l15, ks * 32, lh);
#pragma unroll
    for (int mt = 0; mt < 8; ++mt) {
      AFrag a;
      int e = wave_m * 128 + mt * 16 + l15;
      a.q[0] = *(const uint4*)(Wcur + e * WROW + lh * 8);
      a.q[1] = *(const uint4*)(Wcur + e * WROW + 16 + lh * 8);
      c[mt][0] = wmma_bf16(a, bfr[0], c[mt][0]);
      c[mt][1] = wmma_bf16(a, bfr[1], c[mt][1]);
    }
  }
  // epilogue: bn + relu, bf16 transposed store into Snext [n][d]
#pragma unroll
  for (int mt = 0; mt < 8; ++mt)
#pragma unroll
    for (int nt = 0; nt < 2; ++nt)
#pragma unroll
      for (int v = 0; v < 8; ++v) {
        int e = wave_m * 128 + mt * 16 + v + lh * 8;
        int n = wave_n * 32 + nt * 16 + l15;
        float y = sc[e] * c[mt][nt][v] + sh[e];
        Snext[n * SROW + e] = f2bf(fmaxf(y, 0.f));
      }
  __syncthreads();
}

// acc += gamma * cwB @ softmax(cwT @ S)
__device__ __forceinline__ void vq_accum(const us* Scur,
                                         const us* __restrict__ cwT,
                                         const us* __restrict__ cwB,
                                         int Kj, int Kpad,
                                         const float* __restrict__ gamma,
                                         float* Lg, us* As, v8f (&acc)[8][2],
                                         int tid, int lane, int wid) {
  const int l15 = lane & 15, lh = lane >> 4;
  const int Mt = Kj >> 4;          // 1 / 2 / 4 row tiles of logits
  const int T = Mt * 4;            // x 4 column tiles

  // logits = cwT(Kj x 512) @ S(512 x 64), wave-uniform tile assignment
  for (int t = wid; t < T; t += 8) {
    int mt = t % Mt, nt = t / Mt;
    v8f l = v8zero();
    for (int ks = 0; ks < 16; ++ks) {
      AFrag a, b;
      load_afrag_G(a, cwT, DCH, mt * 16 + l15, ks * 32, lh);
      load_bfrag_S(b, Scur, nt * 16 + l15, ks * 32, lh);
      l = wmma_bf16(a, b, l);
    }
#pragma unroll
    for (int v = 0; v < 8; ++v)
      Lg[(nt * 16 + l15) * LROW + mt * 16 + v + lh * 8] = l[v];
  }
  __syncthreads();

  // per-column softmax (64 columns -> first 64 threads), gamma folded in
  if (tid < NT) {
    float* row = Lg + tid * LROW;
    float m = -3.4e38f;
    for (int k = 0; k < Kj; ++k) m = fmaxf(m, row[k]);
    float s = 0.f;
    for (int k = 0; k < Kj; ++k) { float e = __expf(row[k] - m); row[k] = e; s += e; }
    float g = gamma[0] / s;
    us* ar = As + tid * AROW;
    for (int k = 0; k < Kj; ++k) ar[k] = f2bf(row[k] * g);
    for (int k = Kj; k < Kpad; ++k) ar[k] = 0;
  }
  __syncthreads();

  // acc += cwB(512 x Kpad) @ Asoft(Kpad x 64)
  const int wave_m = wid & 3, wave_n = wid >> 2;
  const int ksteps = Kpad >> 5;
  for (int ks = 0; ks < ksteps; ++ks) {
    AFrag bfr[2];
#pragma unroll
    for (int nt = 0; nt < 2; ++nt) {
      int n = wave_n * 32 + nt * 16 + l15;
      const uint4* p = (const uint4*)(As + n * AROW + ks * 32 + lh * 8);
      bfr[nt].q[0] = p[0];
      bfr[nt].q[1] = p[2];
    }
#pragma unroll
    for (int mt = 0; mt < 8; ++mt) {
      AFrag a;
      load_afrag_G(a, cwB, Kpad, wave_m * 128 + mt * 16 + l15, ks * 32, lh);
      acc[mt][0] = wmma_bf16(a, bfr[0], acc[mt][0]);
      acc[mt][1] = wmma_bf16(a, bfr[1], acc[mt][1]);
    }
  }
  __syncthreads();
}

// ---------------------------------------------------------------------------
// Main fused kernel: one block = (batch b, 64 columns), full D=512 pipeline.
// ---------------------------------------------------------------------------
__global__ __launch_bounds__(256)
void fused_encoding(const float* __restrict__ X, float* __restrict__ out,
                    const us* __restrict__ W1b, const us* __restrict__ W2b,
                    const us* __restrict__ cwT1, const us* __restrict__ cwT2,
                    const us* __restrict__ cwT3,
                    const us* __restrict__ cwB1, const us* __restrict__ cwB2,
                    const us* __restrict__ cwB3,
                    const float* __restrict__ sc1, const float* __restrict__ sh1,
                    const float* __restrict__ sc2, const float* __restrict__ sh2,
                    const float* __restrict__ g1, const float* __restrict__ g2,
                    const float* __restrict__ g3) {
  // LDS carve: S0 | S1 | Wst0 | Wst1 | logits(f32) | Asoft = 236 KB of 320 KB
  __shared__ __align__(16) unsigned char smem[241664];
  us*    S0   = (us*)(smem);
  us*    S1   = (us*)(smem + 66560);
  us*    Wst0 = (us*)(smem + 133120);
  us*    Wst1 = (us*)(smem + 174080);
  float* Lg   = (float*)(smem + 215040);
  us*    As   = (us*)(smem + 232448);

  const int tid = threadIdx.x, lane = tid & 31, wid = tid >> 5;
  const int b = blockIdx.y;
  const int n0 = blockIdx.x * NT;
  const float* Xb = X + (size_t)b * DCH * NSP + n0;

  // stage X tile -> S0 [n][d] bf16 (coalesced f32 reads along n)
  for (int i = tid; i < DCH * NT; i += 256) {
    int n = i & (NT - 1), d = i >> 6;
    S0[n * SROW + d] = f2bf(Xb[(size_t)d * NSP + n]);
  }
  __syncthreads();

  v8f acc[8][2];
#pragma unroll
  for (int mt = 0; mt < 8; ++mt) { acc[mt][0] = v8zero(); acc[mt][1] = v8zero(); }

  vq_accum(S0, cwT1, cwB1, 16, 32, g1, Lg, As, acc, tid, lane, wid);     // E1(X)
  gemm_bn_relu(W1b, S0, S1, Wst0, Wst1, sc1, sh1, tid, lane, wid);       // X2
  vq_accum(S1, cwT2, cwB2, 32, 32, g2, Lg, As, acc, tid, lane, wid);     // E2(X2)
  gemm_bn_relu(W2b, S1, S0, Wst0, Wst1, sc2, sh2, tid, lane, wid);       // X3
  vq_accum(S0, cwT3, cwB3, 64, 64, g3, Lg, As, acc, tid, lane, wid);     // E3(X3)

  // epilogue: out = X + (E1+E2+E3), exact f32 residual from global
  const int l15 = lane & 15, lh = lane >> 4;
  const int wave_m = wid & 3, wave_n = wid >> 2;
#pragma unroll
  for (int mt = 0; mt < 8; ++mt)
#pragma unroll
    for (int nt = 0; nt < 2; ++nt)
#pragma unroll
      for (int v = 0; v < 8; ++v) {
        int d = wave_m * 128 + mt * 16 + v + lh * 8;
        int n = wave_n * 32 + nt * 16 + l15;
        size_t idx = (size_t)(b * DCH + d) * NSP + n0 + n;
        out[idx] = X[idx] + acc[mt][nt][v];
      }
}

// ---------------------------------------------------------------------------
extern "C" void kernel_launch(void* const* d_in, const int* in_sizes, int n_in,
                              void* d_out, int out_size, void* d_ws, size_t ws_size,
                              hipStream_t stream) {
  const float* X   = (const float*)d_in[0];
  const float* W1  = (const float*)d_in[1];
  const float* s1  = (const float*)d_in[2];
  const float* b1  = (const float*)d_in[3];
  const float* m1  = (const float*)d_in[4];
  const float* v1  = (const float*)d_in[5];
  const float* W2  = (const float*)d_in[6];
  const float* s2  = (const float*)d_in[7];
  const float* b2  = (const float*)d_in[8];
  const float* m2  = (const float*)d_in[9];
  const float* v2  = (const float*)d_in[10];
  const float* cw1 = (const float*)d_in[11];
  const float* cw2 = (const float*)d_in[12];
  const float* cw3 = (const float*)d_in[13];
  const float* g1  = (const float*)d_in[14];
  const float* g2  = (const float*)d_in[15];
  const float* g3  = (const float*)d_in[16];
  float* out = (float*)d_out;

  char* w = (char*)d_ws;
  us* W1b  = (us*)(w);                 // 512*512*2 = 524288
  us* W2b  = (us*)(w + 524288);        // 524288
  us* cwT1 = (us*)(w + 1048576);       // 16384
  us* cwT2 = (us*)(w + 1064960);       // 32768
  us* cwT3 = (us*)(w + 1097728);       // 65536
  us* cwB1 = (us*)(w + 1163264);       // 32768
  us* cwB2 = (us*)(w + 1196032);       // 32768
  us* cwB3 = (us*)(w + 1228800);       // 65536
  float* sc1 = (float*)(w + 1294336);  // 2048
  float* sh1 = (float*)(w + 1296384);
  float* sc2 = (float*)(w + 1298432);
  float* sh2 = (float*)(w + 1300480);  // end 1302528 bytes

  prep_kernel<<<512, 256, 0, stream>>>(W1, W2, s1, b1, m1, v1, s2, b2, m2, v2,
                                       cw1, cw2, cw3, W1b, W2b,
                                       cwT1, cwT2, cwT3, cwB1, cwB2, cwB3,
                                       sc1, sh1, sc2, sh2);

  dim3 grid(NSP / NT, 8);              // 144 column tiles x 8 batches
  fused_encoding<<<grid, 256, 0, stream>>>(X, out, W1b, W2b,
                                           cwT1, cwT2, cwT3, cwB1, cwB2, cwB3,
                                           sc1, sh1, sc2, sh2, g1, g2, g3);
}
